// MemTransformerLM_87376814670041
// MI455X (gfx1250) — matrix operand order repaired and checked by reading, not compiled, not run on hardware
//
#include <hip/hip_runtime.h>
#include <hip/hip_bf16.h>

typedef __bf16 v16bf __attribute__((ext_vector_type(16)));
typedef __bf16 v8bf  __attribute__((ext_vector_type(8)));
typedef float  v8f   __attribute__((ext_vector_type(8)));

static __device__ __forceinline__ v16bf cat8(v8bf a, v8bf b) {
  return __builtin_shufflevector(a, b, 0,1,2,3,4,5,6,7,8,9,10,11,12,13,14,15);
}
static __device__ __forceinline__ v8f zero8() {
  v8f z = {0.f,0.f,0.f,0.f,0.f,0.f,0.f,0.f}; return z;
}
static __device__ __forceinline__ v8f wmma_bf16(v16bf a, v16bf b, v8f c) {
#if defined(__HIP_DEVICE_COMPILE__)
  return __builtin_amdgcn_wmma_f32_16x16x32_bf16(false, a, false, b, (short)0, c, false, false);
#else
  return c;
#endif
}
// Build an A-fragment (16x32 bf16) from a per-lane row pointer.
// lanes 0-15: row l, K {0..15}; lanes 16-31: row l, K {16..31}
static __device__ __forceinline__ v16bf ldA(const __bf16* rowp, int kbase, int hi) {
  return *(const v16bf*)(rowp + kbase + hi*16);
}

// ---------------- elementwise helpers ----------------
__global__ __launch_bounds__(256) void k_f2bf(const float* __restrict__ s, __bf16* __restrict__ d, int n) {
  int i = blockIdx.x * 256 + threadIdx.x;
  if (i < n) d[i] = (__bf16)s[i];
}

// cat = [mems ; w]  shape (2048, 2, 1024) flattened -> bf16
__global__ __launch_bounds__(256) void k_cat(const float* __restrict__ mems, const float* __restrict__ w,
                                             __bf16* __restrict__ cat) {
  int i = blockIdx.x * 256 + threadIdx.x;            // 4,194,304 elems
  float v = (i < 2097152) ? mems[i] : w[i - 2097152];
  cat[i] = (__bf16)v;
}

// unpack w_heads (k, b, 3072) -> per-(b,n) bf16 buffers
__global__ __launch_bounds__(256) void k_pack(const float* __restrict__ wh, const float* __restrict__ rwb,
                                              __bf16* __restrict__ kk, __bf16* __restrict__ vt,
                                              __bf16* __restrict__ qrw, __bf16* __restrict__ qn) {
  int idx = blockIdx.x * 256 + threadIdx.x;          // 2048*2*16*64
  int d = idx & 63, n = (idx >> 6) & 15, b = (idx >> 10) & 1, j = idx >> 11;
  size_t whrow = ((size_t)j * 2 + b) * 3072;
  int bn = b * 16 + n;
  float kvk = wh[whrow + 1024 + n * 64 + d];
  kk[((size_t)bn * 2048 + j) * 64 + d] = (__bf16)kvk;
  float vv = wh[whrow + 2048 + n * 64 + d];
  vt[((size_t)bn * 64 + d) * 2048 + j] = (__bf16)vv;   // transposed V for PV B-fragments
  if (j >= 1024) {
    int i = j - 1024;
    float qv = wh[whrow + n * 64 + d];
    qrw[((size_t)bn * 1024 + i) * 64 + d] = (__bf16)(qv + rwb[n * 64 + d]);
    qn [((size_t)bn * 1024 + i) * 64 + d] = (__bf16)qv;
  }
}

// ---------------- generic bf16 WMMA GEMM: C[M,N] = A[M,K] * B[N,K]^T ----------------
// Block tile 256(M) x 128(N), 8 waves, each wave computes 32x128.
// Double-buffered LDS, one barrier per K-step, global prefetch 2 steps ahead.
__global__ __launch_bounds__(256) void k_gemm(
    const __bf16* __restrict__ A, const __bf16* __restrict__ B,
    float* __restrict__ Cf, __bf16* __restrict__ Cb,
    const float* __restrict__ bias, int M, int N, int K, int relu)
{
  __shared__ __align__(32) __bf16 As[2][256 * 32];
  __shared__ __align__(32) __bf16 Bs[2][128 * 32];
  int tid = threadIdx.x;
  int wave = tid >> 5, lane = tid & 31, l = lane & 15, hi = lane >> 4;
  int m0 = blockIdx.y * 256, n0 = blockIdx.x * 128;

  v8f acc[2][8];
#pragma unroll
  for (int ms = 0; ms < 2; ++ms)
#pragma unroll
    for (int ct = 0; ct < 8; ++ct) acc[ms][ct] = zero8();

  // staging assignment: thread t loads A row (m0+t) 32 elems; B row (n0 + t>>1), 16-elem half (t&1)
  const __bf16* Aptr = A + (size_t)(m0 + tid) * K;
  const __bf16* Bptr = B + (size_t)(n0 + (tid >> 1)) * K + (tid & 1) * 16;
  int bofs = (tid >> 1) * 32 + (tid & 1) * 16;

  v16bf ra0, ra1, rb;
  int nst = K >> 5;
  // prologue: fetch step0, stage buf0, fetch step1
  ra0 = *(const v16bf*)(Aptr + 0);
  ra1 = *(const v16bf*)(Aptr + 16);
  rb  = *(const v16bf*)(Bptr + 0);
  *(v16bf*)&As[0][tid * 32]      = ra0;
  *(v16bf*)&As[0][tid * 32 + 16] = ra1;
  *(v16bf*)&Bs[0][bofs]          = rb;
  ra0 = *(const v16bf*)(Aptr + 32);
  ra1 = *(const v16bf*)(Aptr + 48);
  rb  = *(const v16bf*)(Bptr + 32);
  __syncthreads();

  for (int s = 0; s < nst; ++s) {
    int cur = s & 1;
    if (s + 1 < nst) {                       // stage next buffer (idle since barrier of step s-1)
      *(v16bf*)&As[cur ^ 1][tid * 32]      = ra0;
      *(v16bf*)&As[cur ^ 1][tid * 32 + 16] = ra1;
      *(v16bf*)&Bs[cur ^ 1][bofs]          = rb;
    }
    if (s + 2 < nst) {                       // prefetch 2 steps ahead into registers
      int k0 = (s + 2) << 5;
      ra0 = *(const v16bf*)(Aptr + k0);
      ra1 = *(const v16bf*)(Aptr + k0 + 16);
      rb  = *(const v16bf*)(Bptr + k0);
    }
    // compute on current buffer
    v16bf af[2];
#pragma unroll
    for (int ms = 0; ms < 2; ++ms) {
      int arow = (wave * 32 + ms * 16 + l) * 32;
      af[ms] = *(const v16bf*)&As[cur][arow + hi * 16];
    }
#pragma unroll
    for (int ct = 0; ct < 8; ++ct) {
      v16bf bfrag = *(const v16bf*)&Bs[cur][(ct * 16 + l) * 32 + hi * 16];
      acc[0][ct] = wmma_bf16(af[0], bfrag, acc[0][ct]);
      acc[1][ct] = wmma_bf16(af[1], bfrag, acc[1][ct]);
    }
    __syncthreads();
  }

#pragma unroll
  for (int ct = 0; ct < 8; ++ct) {
    int col = n0 + ct * 16 + l;
    float bv = bias ? bias[col] : 0.f;
#pragma unroll
    for (int ms = 0; ms < 2; ++ms) {
#pragma unroll
      for (int r = 0; r < 8; ++r) {
        int row = m0 + wave * 32 + ms * 16 + hi * 8 + r;
        float v = acc[ms][ct][r] + bv;
        if (relu) v = fmaxf(v, 0.f);
        size_t o = (size_t)row * N + col;
        if (Cf) Cf[o] = v;
        if (Cb) Cb[o] = (__bf16)v;
      }
    }
  }
}

// ---------------- fused rel-pos attention (flash-style), one (b,n,i-tile16) per block ----------------
__global__ __launch_bounds__(256) void k_attn(
    const __bf16* __restrict__ qrw, const __bf16* __restrict__ qn,
    const __bf16* __restrict__ kk,  const __bf16* __restrict__ vt,
    const __bf16* __restrict__ remb, const float* __restrict__ rbias,
    __bf16* __restrict__ attn_vec)
{
  __shared__ float S[16 * 256];          // AC scores; assembled scores stay in sreg
  __shared__ float Bt[16 * 272];         // unshifted Q.R^T band
  __shared__ __align__(32) __bf16 P[16 * 256];
  __shared__ float red[256];
  __shared__ float rowm[16], rowl[16], rowf[16];
  __shared__ float Obuf[16 * 64];

  int tid = threadIdx.x;
  int wave = tid >> 5, lane = tid & 31, l = lane & 15, hi = lane >> 4;
  int bid = blockIdx.x;
  int it = bid & 63, bn = bid >> 6;
  int b = bn >> 4, n = bn & 15;
  int i0 = it * 16;

  const __bf16* qrwB = qrw + ((size_t)bn * 1024 + i0 + l) * 64;
  const __bf16* qnB  = qn  + ((size_t)bn * 1024 + i0 + l) * 64;
  v16bf aQrw0 = ldA(qrwB, 0, hi), aQrw1 = ldA(qrwB, 32, hi);
  v16bf aQ0   = ldA(qnB, 0, hi),  aQ1   = ldA(qnB, 32, hi);

  if (tid < 16) { rowm[tid] = -1e30f; rowl[tid] = 0.f; }
  __syncthreads();

  v8f oacc = zero8();
  int dt = wave & 3, par = wave >> 2;
  const float scale = 0.125f;
  int ii = tid >> 4, cb = (tid & 15) * 16;

  for (int ch = 0; ch < 8; ++ch) {
    int j0 = ch * 256;
    int jjbase = j0 + 1008 - i0;
    // --- AC tiles (16 tiles of 16x16, K=64) ---
#pragma unroll
    for (int tt = 0; tt < 2; ++tt) {
      int t = wave * 2 + tt;
      int jr = j0 + t * 16 + l;
      const __bf16* kB = kk + ((size_t)bn * 2048 + jr) * 64;
      v16bf b0 = *(const v16bf*)(kB + hi * 16);
      v16bf b1 = *(const v16bf*)(kB + 32 + hi * 16);
      v8f a = zero8();
      a = wmma_bf16(aQrw0, b0, a);
      a = wmma_bf16(aQrw1, b1, a);
#pragma unroll
      for (int r = 0; r < 8; ++r) S[(hi * 8 + r) * 256 + t * 16 + l] = a[r];
    }
    // --- Q.R^T band tiles (17 tiles) ---
    for (int t = wave; t < 17; t += 8) {
      int jj = jjbase + t * 16 + l; if (jj > 2047) jj = 2047;
      const __bf16* rB = remb + ((size_t)jj * 16 + n) * 64;
      v16bf b0 = *(const v16bf*)(rB + hi * 16);
      v16bf b1 = *(const v16bf*)(rB + 32 + hi * 16);
      v8f a = zero8();
      a = wmma_bf16(aQ0, b0, a);
      a = wmma_bf16(aQ1, b1, a);
#pragma unroll
      for (int r = 0; r < 8; ++r) Bt[(hi * 8 + r) * 272 + t * 16 + l] = a[r];
    }
    __syncthreads();
    // --- assemble scores: AC + shifted(B_+D_) , mask, scale ---
    int i = i0 + ii;
    float vmax = -1e30f;
    float sreg[16];
#pragma unroll
    for (int c = 0; c < 16; ++c) {
      int j = j0 + cb + c;
      float s;
      if (j > i + 1024) s = -1e30f;
      else {
        int jj = j + 1023 - i;
        s = (S[ii * 256 + cb + c] + Bt[ii * 272 + (jj - jjbase)] + rbias[jj * 16 + n]) * scale;
      }
      sreg[c] = s; vmax = fmaxf(vmax, s);
    }
    red[tid] = vmax;
    __syncthreads();
    if (!(tid & 15)) {
      float cm = red[tid];
      for (int u = 1; u < 16; ++u) cm = fmaxf(cm, red[tid + u]);
      float mo = rowm[ii], mn = fmaxf(mo, cm);
      rowm[ii] = mn; rowf[ii] = __expf(mo - mn);
    }
    __syncthreads();
    float mn = rowm[ii], ps = 0.f;
#pragma unroll
    for (int c = 0; c < 16; ++c) {
      float p = __expf(sreg[c] - mn);
      ps += p;
      P[ii * 256 + cb + c] = (__bf16)p;
    }
    red[tid] = ps;
    __syncthreads();
    if (!(tid & 15)) {
      float s2 = 0.f;
      for (int u = 0; u < 16; ++u) s2 += red[tid + u];
      rowl[ii] = rowl[ii] * rowf[ii] + s2;
    }
    __syncthreads();
    // --- rescale running output, accumulate P * V for this chunk ---
#pragma unroll
    for (int r = 0; r < 8; ++r) oacc[r] *= rowf[hi * 8 + r];
    const __bf16* vB = vt + ((size_t)bn * 64 + dt * 16 + l) * 2048 + j0;
#pragma unroll
    for (int k2 = 0; k2 < 4; ++k2) {
      int kt = k2 * 2 + par;
      v16bf pa = *(const v16bf*)&P[l * 256 + kt * 32 + hi * 16];
      v16bf vb = *(const v16bf*)(vB + kt * 32 + hi * 16);
      oacc = wmma_bf16(pa, vb, oacc);
    }
    __syncthreads();
  }
  // combine the two k-parity halves, normalize, write bf16 attn_vec
#pragma unroll
  for (int r = 0; r < 8; ++r) if (!par) Obuf[(hi * 8 + r) * 64 + dt * 16 + l] = oacc[r];
  __syncthreads();
#pragma unroll
  for (int r = 0; r < 8; ++r) if (par) Obuf[(hi * 8 + r) * 64 + dt * 16 + l] += oacc[r];
  __syncthreads();
  for (int e = tid; e < 1024; e += 256) {
    int ri = e >> 6, d = e & 63;
    float v = Obuf[e] / rowl[ri];
    attn_vec[((size_t)((i0 + ri) * 2 + b)) * 1024 + n * 64 + d] = (__bf16)v;
  }
}

// ---------------- residual + LayerNorm over 1024, one row per block ----------------
__global__ __launch_bounds__(256) void k_ln(const float* __restrict__ res, const float* __restrict__ y,
                                            const float* __restrict__ g, const float* __restrict__ bb,
                                            float* __restrict__ outF, __bf16* __restrict__ outB)
{
  __shared__ float r1[256], r2[256];
  int row = blockIdx.x, tid = threadIdx.x;
  const float* xr = res + (size_t)row * 1024;
  const float* yr = y + (size_t)row * 1024;
  float x[4]; float s = 0.f, s2 = 0.f;
#pragma unroll
  for (int u = 0; u < 4; ++u) {
    int c = tid + u * 256;
    float v = xr[c] + yr[c];
    x[u] = v; s += v; s2 += v * v;
  }
  r1[tid] = s; r2[tid] = s2; __syncthreads();
  for (int off = 128; off; off >>= 1) {
    if (tid < off) { r1[tid] += r1[tid + off]; r2[tid] += r2[tid + off]; }
    __syncthreads();
  }
  float mean = r1[0] * (1.f / 1024.f);
  float var = r2[0] * (1.f / 1024.f) - mean * mean;
  float rs = rsqrtf(var + 1e-5f);
#pragma unroll
  for (int u = 0; u < 4; ++u) {
    int c = tid + u * 256;
    float o = g[c] * (x[u] - mean) * rs + bb[c];
    if (outF) outF[(size_t)row * 1024 + c] = o;
    if (outB) outB[(size_t)row * 1024 + c] = (__bf16)o;
  }
}

extern "C" void kernel_launch(void* const* d_in, const int* in_sizes, int n_in,
                              void* d_out, int out_size, void* d_ws, size_t ws_size,
                              hipStream_t stream) {
  (void)in_sizes; (void)n_in; (void)out_size; (void)ws_size;
  const float* w       = (const float*)d_in[0];
  const float* mems    = (const float*)d_in[1];
  const float* r_emb   = (const float*)d_in[2];
  const float* r_w_bias= (const float*)d_in[3];
  const float* r_bias  = (const float*)d_in[4];
  const float* qkv_w   = (const float*)d_in[5];
  const float* o_w     = (const float*)d_in[6];
  const float* ln1_g   = (const float*)d_in[7];
  const float* ln1_b   = (const float*)d_in[8];
  const float* ff_w1   = (const float*)d_in[9];
  const float* ff_b1   = (const float*)d_in[10];
  const float* ff_w2   = (const float*)d_in[11];
  const float* ff_b2   = (const float*)d_in[12];
  const float* ln2_g   = (const float*)d_in[13];
  const float* ln2_b   = (const float*)d_in[14];

  char* p = (char*)d_ws;
  auto carve = [&](size_t bytes) -> void* {
    char* r = p; p += (bytes + 255) & ~(size_t)255; return (void*)r;
  };
  __bf16* cat_bf   = (__bf16*)carve(4096ull * 1024 * 2);
  __bf16* qkvw_bf  = (__bf16*)carve(3072ull * 1024 * 2);
  __bf16* remb_bf  = (__bf16*)carve(2048ull * 16 * 64 * 2);
  __bf16* ow_bf    = (__bf16*)carve(1024ull * 1024 * 2);
  __bf16* ffw1_bf  = (__bf16*)carve(4096ull * 1024 * 2);
  __bf16* ffw2_bf  = (__bf16*)carve(1024ull * 4096 * 2);
  float*  wheads   = (float*) carve(4096ull * 3072 * 4);
  __bf16* kk_bf    = (__bf16*)carve(32ull * 2048 * 64 * 2);
  __bf16* vt_bf    = (__bf16*)carve(32ull * 64 * 2048 * 2);
  __bf16* qrw_bf   = (__bf16*)carve(32ull * 1024 * 64 * 2);
  __bf16* qn_bf    = (__bf16*)carve(32ull * 1024 * 64 * 2);
  __bf16* avec_bf  = (__bf16*)carve(2048ull * 1024 * 2);
  float*  attnout  = (float*) carve(2048ull * 1024 * 4);
  float*  h_f32    = (float*) carve(2048ull * 1024 * 4);
  __bf16* h_bf     = (__bf16*)carve(2048ull * 1024 * 2);
  __bf16* inner_bf = (__bf16*)carve(2048ull * 4096 * 2);
  float*  ff_f32   = (float*) carve(2048ull * 1024 * 4);

  // precision conversions
  k_cat <<<16384, 256, 0, stream>>>(mems, w, cat_bf);
  k_f2bf<<<12288, 256, 0, stream>>>(qkv_w, qkvw_bf, 3072 * 1024);
  k_f2bf<<< 8192, 256, 0, stream>>>(r_emb, remb_bf, 2048 * 16 * 64);
  k_f2bf<<< 4096, 256, 0, stream>>>(o_w, ow_bf, 1024 * 1024);
  k_f2bf<<<16384, 256, 0, stream>>>(ff_w1, ffw1_bf, 4096 * 1024);
  k_f2bf<<<16384, 256, 0, stream>>>(ff_w2, ffw2_bf, 1024 * 4096);
  // QKV projection: (4096 x 1024) * (3072 x 1024)^T
  k_gemm<<<dim3(24, 16), 256, 0, stream>>>(cat_bf, qkvw_bf, wheads, nullptr, nullptr, 4096, 3072, 1024, 0);
  // unpack q(+bias)/k/v into per-(b,n) bf16 layouts (V transposed)
  k_pack<<<16384, 256, 0, stream>>>(wheads, r_w_bias, kk_bf, vt_bf, qrw_bf, qn_bf);
  // fused rel-pos attention -> attn_vec bf16
  k_attn<<<2048, 256, 0, stream>>>(qrw_bf, qn_bf, kk_bf, vt_bf, remb_bf, r_bias, avec_bf);
  // output projection
  k_gemm<<<dim3(8, 8), 256, 0, stream>>>(avec_bf, ow_bf, attnout, nullptr, nullptr, 2048, 1024, 1024, 0);
  // h = LN(w + attn_out)
  k_ln<<<2048, 256, 0, stream>>>(w, attnout, ln1_g, ln1_b, h_f32, h_bf);
  // inner = relu(h * ff_w1^T + b1)  -> bf16
  k_gemm<<<dim3(32, 8), 256, 0, stream>>>(h_bf, ffw1_bf, nullptr, inner_bf, ff_b1, 2048, 4096, 1024, 1);
  // ff_out = inner * ff_w2^T + b2
  k_gemm<<<dim3(8, 8), 256, 0, stream>>>(inner_bf, ffw2_bf, ff_f32, nullptr, ff_b2, 2048, 1024, 4096, 0);
  // out = LN(h + ff_out)
  k_ln<<<2048, 256, 0, stream>>>(h_f32, ff_f32, ln2_g, ln2_b, (float*)d_out, nullptr);
}
